// FixedPointBitLinear_12386685681974
// MI455X (gfx1250) — 8-bit matrix operand order repaired and checked by reading, compiled, not run on hardware
//
#include <hip/hip_runtime.h>


typedef int v8i __attribute__((ext_vector_type(8)));

#define DIN   4096
#define DOUT  4096
#define MROWS 4096
#define PLANE_BYTES ((size_t)MROWS * DIN)

static __device__ __forceinline__ v8i wmma_iu8_u(v8i a, v8i b, v8i c) {
  return __builtin_amdgcn_wmma_i32_16x16x64_iu8(false, a, true, b, c, false, false);
}
static __device__ __forceinline__ v8i wmma_iu8_s(v8i a, v8i b, v8i c) {
  return __builtin_amdgcn_wmma_i32_16x16x64_iu8(true, a, true, b, c, false, false);
}

static __device__ __forceinline__ v8i load_a(const unsigned char* __restrict__ p,
                                             int mbase, int kc, int half, int lN) {
  const unsigned char* q = p + (size_t)(mbase + lN) * DIN + (kc + half * 16);
  uint4 q0 = *(const uint4*)(q);
  uint4 q1 = *(const uint4*)(q + 32);
  v8i a;
  a[0] = (int)q0.x; a[1] = (int)q0.y; a[2] = (int)q0.z; a[3] = (int)q0.w;
  a[4] = (int)q1.x; a[5] = (int)q1.y; a[6] = (int)q1.z; a[7] = (int)q1.w;
  return a;
}

static __device__ __forceinline__ v8i load_b(const unsigned char* __restrict__ w8,
                                             int nbase, int kc, int half, int lN) {
  const unsigned char* q = w8 + (size_t)(nbase + lN) * DIN + (kc + half * 16);
  uint4 r0 = *(const uint4*)(q);
  uint4 r1 = *(const uint4*)(q + 32);
  v8i b;
  b[0] = (int)r0.x; b[1] = (int)r0.y; b[2] = (int)r0.z; b[3] = (int)r0.w;
  b[4] = (int)r1.x; b[5] = (int)r1.y; b[6] = (int)r1.z; b[7] = (int)r1.w;
  return b;
}

__global__ void bl_init(long long* scal) { scal[0] = 0; }

__global__ __launch_bounds__(256) void bl_wabs_sum(const float* __restrict__ w,
                                                   unsigned long long* __restrict__ sum) {
  __shared__ long long s[256];
  const int t = threadIdx.x;
  const size_t base = (size_t)blockIdx.x * 4096 + (size_t)t * 16;
  long long acc = 0;
#pragma unroll
  for (int i = 0; i < 16; ++i) {
    int v = (int)rintf(w[base + i] * 65536.0f);
    acc += (v < 0) ? -(long long)v : (long long)v;
  }
  s[t] = acc;
  __syncthreads();
  for (int off = 128; off > 0; off >>= 1) {
    if (t < off) s[t] += s[t + off];
    __syncthreads();
  }
  if (t == 0) atomicAdd(sum, (unsigned long long)s[0]);
}

__global__ void bl_wfinal(long long* scal) {
  long long mean = scal[0] >> 24;
  if (mean < 1) mean = 1;
  scal[1] = mean;
  scal[2] = ((long long)1 << 32) / mean;
}

__global__ __launch_bounds__(256) void bl_wtern(const float* __restrict__ w,
                                                const long long* __restrict__ scal,
                                                unsigned char* __restrict__ w8) {
  const long long mean = scal[1];
  const size_t i0 = ((size_t)blockIdx.x * 256 + threadIdx.x) * 4;
  unsigned pack = 0;
#pragma unroll
  for (int j = 0; j < 4; ++j) {
    int v = (int)rintf(w[i0 + j] * 65536.0f);
    int tern = (v >= mean) ? 1 : ((v < 0) ? -1 : 0);
    pack |= ((unsigned)(unsigned char)(signed char)tern) << (j * 8);
  }
  *(unsigned*)(w8 + i0) = pack;
}

__global__ __launch_bounds__(256) void bl_actquant(const float* __restrict__ x,
                                                   unsigned char* __restrict__ lo,
                                                   unsigned char* __restrict__ mid,
                                                   unsigned char* __restrict__ hi) {
  __shared__ int smax[256];
  __shared__ long long sc[2];
  const int row = blockIdx.x;
  const int t = threadIdx.x;
  const float* xr = x + (size_t)row * DIN + (size_t)t * 16;

  int xf[16];
  int mx = 0;
#pragma unroll
  for (int i = 0; i < 16; ++i) {
    int v = (int)rintf(xr[i] * 65536.0f);
    xf[i] = v;
    int a = (v < 0) ? -v : v;
    mx = (a > mx) ? a : mx;
  }
  smax[t] = mx;
  __syncthreads();
  for (int off = 128; off > 0; off >>= 1) {
    if (t < off) { int o = smax[t + off]; if (o > smax[t]) smax[t] = o; }
    __syncthreads();
  }
  if (t == 0) {
    long long m = smax[0];
    if (m < 1) m = 1;
    long long scale = ((long long)127 << 32) / m;
    sc[0] = scale;
    sc[1] = ((long long)1 << 32) / scale;
  }
  __syncthreads();
  const long long scale = sc[0], sinv = sc[1];

  unsigned pl[4] = {0, 0, 0, 0}, pm[4] = {0, 0, 0, 0}, ph[4] = {0, 0, 0, 0};
#pragma unroll
  for (int i = 0; i < 16; ++i) {
    long long xs = ((long long)xf[i] * scale) >> 16;
    if (xs < -8388608LL) xs = -8388608LL;
    if (xs >  8323072LL) xs =  8323072LL;
    long long q = (xs * sinv) >> 16;
    unsigned u = (unsigned)q & 0xFFFFFFu;
    unsigned sh = (unsigned)((i & 3) * 8);
    pl[i >> 2] |= (u & 0xFFu) << sh;
    pm[i >> 2] |= ((u >> 8) & 0xFFu) << sh;
    ph[i >> 2] |= ((u >> 16) & 0xFFu) << sh;
  }
  const size_t base = (size_t)row * DIN + (size_t)t * 16;
  *(uint4*)(lo  + base) = make_uint4(pl[0], pl[1], pl[2], pl[3]);
  *(uint4*)(mid + base) = make_uint4(pm[0], pm[1], pm[2], pm[3]);
  *(uint4*)(hi  + base) = make_uint4(ph[0], ph[1], ph[2], ph[3]);
}

__global__ __launch_bounds__(256) void bl_wmma_gemm(
    const unsigned char* __restrict__ lo,
    const unsigned char* __restrict__ mid,
    const unsigned char* __restrict__ hi,
    const unsigned char* __restrict__ w8,
    const long long* __restrict__ scal,
    float* __restrict__ out) {
  const int lane = threadIdx.x & 31;
  const int wave = threadIdx.x >> 5;
  const int half = lane >> 4;
  const int lN = lane & 15;
  const int m0 = blockIdx.y * 128 + (wave >> 1) * 32;
  const int n0 = blockIdx.x * 128 + (wave & 1) * 64;

  v8i acc[3][2][4] = {};

#pragma unroll 1
  for (int kc = 0; kc < DIN; kc += 64) {
    v8i Bv[4];
#pragma unroll
    for (int nt = 0; nt < 4; ++nt) Bv[nt] = load_b(w8, n0 + nt * 16, kc, half, lN);
    if (kc + 64 < DIN)
      __builtin_prefetch(w8 + (size_t)(n0 + lane) * DIN + kc + 64, 0, 1);

    v8i A;
    A = load_a(lo, m0, kc, half, lN);
#pragma unroll
    for (int nt = 0; nt < 4; ++nt) acc[0][0][nt] = wmma_iu8_u(A, Bv[nt], acc[0][0][nt]);
    A = load_a(lo, m0 + 16, kc, half, lN);
#pragma unroll
    for (int nt = 0; nt < 4; ++nt) acc[0][1][nt] = wmma_iu8_u(A, Bv[nt], acc[0][1][nt]);
    A = load_a(mid, m0, kc, half, lN);
#pragma unroll
    for (int nt = 0; nt < 4; ++nt) acc[1][0][nt] = wmma_iu8_u(A, Bv[nt], acc[1][0][nt]);
    A = load_a(mid, m0 + 16, kc, half, lN);
#pragma unroll
    for (int nt = 0; nt < 4; ++nt) acc[1][1][nt] = wmma_iu8_u(A, Bv[nt], acc[1][1][nt]);
    A = load_a(hi, m0, kc, half, lN);
#pragma unroll
    for (int nt = 0; nt < 4; ++nt) acc[2][0][nt] = wmma_iu8_s(A, Bv[nt], acc[2][0][nt]);
    A = load_a(hi, m0 + 16, kc, half, lN);
#pragma unroll
    for (int nt = 0; nt < 4; ++nt) acc[2][1][nt] = wmma_iu8_s(A, Bv[nt], acc[2][1][nt]);
  }

  const long long sw = scal[2];
#pragma unroll
  for (int mt = 0; mt < 2; ++mt) {
#pragma unroll
    for (int nt = 0; nt < 4; ++nt) {
      const int Nc = n0 + nt * 16 + lN;
      const int Mb = m0 + mt * 16 + half * 8;
#pragma unroll
      for (int j = 0; j < 8; ++j) {
        unsigned a32 = (unsigned)acc[0][mt][nt][j]
                     + ((unsigned)acc[1][mt][nt][j] << 8)
                     + ((unsigned)acc[2][mt][nt][j] << 16);
        long long tq = ((long long)(int)a32 * sw) >> 16;
        out[(size_t)(Mb + j) * DOUT + Nc] = (float)tq * (1.0f / 65536.0f);
      }
    }
  }
}

extern "C" void kernel_launch(void* const* d_in, const int* in_sizes, int n_in,
                              void* d_out, int out_size, void* d_ws, size_t ws_size,
                              hipStream_t stream) {
  (void)in_sizes; (void)n_in; (void)out_size; (void)ws_size;
  const float* x = (const float*)d_in[0];
  const float* w = (const float*)d_in[1];
  float* out = (float*)d_out;

  char* ws = (char*)d_ws;
  long long* scal        = (long long*)ws;
  unsigned char* w8      = (unsigned char*)(ws + 256);
  unsigned char* lo      = w8 + PLANE_BYTES;
  unsigned char* mid     = lo + PLANE_BYTES;
  unsigned char* hi      = mid + PLANE_BYTES;

  bl_init<<<1, 1, 0, stream>>>(scal);
  bl_wabs_sum<<<4096, 256, 0, stream>>>(w, (unsigned long long*)scal);
  bl_wfinal<<<1, 1, 0, stream>>>(scal);
  bl_wtern<<<16384, 256, 0, stream>>>(w, scal, w8);
  bl_actquant<<<MROWS, 256, 0, stream>>>(x, lo, mid, hi);

  dim3 grid(DOUT / 128, MROWS / 128);
  bl_wmma_gemm<<<grid, 256, 0, stream>>>(lo, mid, hi, w8, scal, out);
}
